// TemporalLSTM_67233418051655
// MI455X (gfx1250) — matrix operand order repaired and checked by reading, not compiled, not run on hardware
//
#include <hip/hip_runtime.h>
#include <stdint.h>
#include <stddef.h>

// Problem dims (fixed by the reference)
#define D_IN   512
#define H_DIM  2048
#define T_SEQ  2048
#define O_DIM  512
#define FOURH  8192

// ---- vector types ----
typedef __attribute__((ext_vector_type(16))) __bf16       v16bf;
typedef __attribute__((ext_vector_type(8)))  __bf16       v8bf;
typedef __attribute__((ext_vector_type(8)))  float        v8f;
typedef __attribute__((ext_vector_type(4)))  unsigned int v4u;
typedef __attribute__((ext_vector_type(2)))  float        v2f;

__device__ __forceinline__ unsigned short f32_to_bf16(float f) {
  unsigned u = __float_as_uint(f);
  u += 0x7fffu + ((u >> 16) & 1u);          // round-to-nearest-even
  return (unsigned short)(u >> 16);
}
__device__ __forceinline__ float bf16lo(unsigned u) { return __uint_as_float(u << 16); }
__device__ __forceinline__ float bf16hi(unsigned u) { return __uint_as_float(u & 0xffff0000u); }

// ---------------------------------------------------------------------------
// fp32 -> bf16 conversion (grid-stride)
// ---------------------------------------------------------------------------
__global__ __launch_bounds__(256) void f2bf_kernel(const float* __restrict__ s,
                                                   unsigned short* __restrict__ d,
                                                   size_t n) {
  size_t i  = (size_t)blockIdx.x * blockDim.x + threadIdx.x;
  size_t st = (size_t)gridDim.x * blockDim.x;
  for (; i < n; i += st) d[i] = f32_to_bf16(s[i]);
}

__global__ void zero_bar_kernel(unsigned* bar) { *bar = 0u; }

// ---------------------------------------------------------------------------
// WMMA bf16 GEMM:  C[m][n] = sum_k A[m][k] * B[n][k] + bias_a[n] + bias_b[n]
// (A: M x K row-major, B: N x K row-major -> computes A @ B^T, exactly the
//  x @ Wih^T shape of the reference.)  One wave per 16x16 C tile, K-step 32.
// CDNA5 fragment layouts as used here (A/B share one contiguous k order; C per ISA 7.12.2):
//  A (16x32 bf16): lane = M (l&15); elems i -> K = k0 + (l>>4)*16 + i
//  B (32x16 bf16): lane = N (l&15); elems i -> K = k0 + (l>>4)*16 + i
//  C (16x16 f32):  vgpr r, M = r + (l>>4)*8, N = l&15
// ---------------------------------------------------------------------------
__global__ __launch_bounds__(256) void gemm_bf16_wmma(
    const unsigned short* __restrict__ A, int lda,
    const unsigned short* __restrict__ B, int ldb,
    const float* __restrict__ bias_a, const float* __restrict__ bias_b,
    float* __restrict__ C, int M, int N, int K)
{
  const int gwave = (int)((blockIdx.x * blockDim.x + threadIdx.x) >> 5);
  const int lane  = (int)(threadIdx.x & 31);
  const int ntn   = N >> 4;
  const int tm    = gwave / ntn;
  const int tn    = gwave - tm * ntn;
  if (tm >= (M >> 4)) return;              // wave-uniform: EXEC stays all-ones

  const int lanelo = lane & 15;
  const int khalfA = (lane >> 4) << 4;     // 0 or 16
  const int khalfB = (lane >> 4) << 4;     // 0 or 16

  const unsigned short* arow = A + (size_t)((tm << 4) + lanelo) * lda;
  const unsigned short* brow = B + (size_t)((tn << 4) + lanelo) * ldb;

  v8f acc = {};
  for (int k0 = 0; k0 < K; k0 += 32) {
    const unsigned short* pa = arow + k0 + khalfA;
    v8bf alo = *(const v8bf*)(pa);
    v8bf ahi = *(const v8bf*)(pa + 8);
    v16bf af = __builtin_shufflevector(alo, ahi, 0,1,2,3,4,5,6,7,8,9,10,11,12,13,14,15);

    const unsigned short* pb = brow + k0 + khalfB;
    v8bf blo = *(const v8bf*)(pb);
    v8bf bhi = *(const v8bf*)(pb + 8);
    v16bf bf = __builtin_shufflevector(blo, bhi, 0,1,2,3,4,5,6,7,8,9,10,11,12,13,14,15);

    acc = __builtin_amdgcn_wmma_f32_16x16x32_bf16(false, af, false, bf,
                                                  (short)0, acc, false, false);
  }

  const int n     = (tn << 4) + lanelo;
  const float bv  = bias_a[n] + bias_b[n];
  const int mbase = (tm << 4) + ((lane >> 4) << 3);
  float* crow = C + (size_t)mbase * N + n;
#pragma unroll
  for (int r = 0; r < 8; ++r) crow[(size_t)r * N] = acc[r] + bv;
}

// ---------------------------------------------------------------------------
// Persistent LSTM recurrence, Whh LDS-resident.
//  - RG=128 workgroups x 256 threads; WG b owns hidden units [16b, 16b+16).
//  - LDS: 64 rows (4 gates x 16 units) x 2048 bf16, row padded by 16B so that
//    ds_load_b128 across a wave hits 64 distinct banks (conflict-free).
//  - Each step: cache this lane's 64-element h slice (global bf16, L2
//    broadcast), 8 rows/wave dot products with v2f accumulate, wave shfl
//    reduce, then 16 lanes do the gate nonlinearities + cell update.
//  - Device-scope barrier per step (monotonic counter, release/acquire at
//    agent scope).  h is double-buffered so one barrier per step suffices.
//  NOTE: assumes the 128 WGs (256KB LDS each -> 1/WGP) are co-resident,
//  which holds on an MI455X-class part (>=128 WGPs).
// ---------------------------------------------------------------------------
#define RG      128
#define RB      256
#define RU      16
#define RROWS   64
#define ROWU16  2056                       // 2048 + 8 bf16 pad (16 bytes)
#define LW_BYTES (RROWS * ROWU16 * 2)
#define RSMEM    (LW_BYTES + RROWS * 4 + RU * 4)

__global__ __launch_bounds__(RB) void lstm_recurrence(
    const float* __restrict__ Whh,        // (4H, H) fp32
    const float* __restrict__ xproj,      // (T, 4H) fp32 (precomputed x@Wih^T + b)
    unsigned short* __restrict__ ys,      // (T, H) bf16 out
    unsigned short* __restrict__ hbuf,    // 2 x H bf16 (double buffer)
    float* __restrict__ hf32,             // H fp32 (final h)
    unsigned* __restrict__ bar, int T)
{
  extern __shared__ char smem[];
  unsigned short* lw   = (unsigned short*)smem;
  float*          zsum = (float*)(smem + LW_BYTES);
  float*          cst  = zsum + RROWS;

  const int tid  = (int)threadIdx.x;
  const int lane = tid & 31;
  const int wid  = tid >> 5;
  const int u0   = (int)blockIdx.x * RU;

  // Preload + convert this WG's Whh slice into LDS (read fp32 once).
  for (int idx = tid; idx < RROWS * H_DIM; idx += RB) {
    int r = idx >> 11;                     // / 2048
    int k = idx & (H_DIM - 1);
    int gate = r >> 4, u = r & 15;
    float wv = Whh[(size_t)(gate * H_DIM + u0 + u) * H_DIM + k];
    lw[r * ROWU16 + k] = f32_to_bf16(wv);
  }
  if (tid < RU) {
    cst[tid] = 0.0f;
    hbuf[u0 + tid] = 0;                    // h(-1) = 0, both buffers
    hbuf[H_DIM + u0 + tid] = 0;
  }

  unsigned barcount = 0;
#define GRID_BARRIER() do {                                                    \
    __syncthreads();                                                           \
    if (tid == 0) {                                                            \
      __hip_atomic_fetch_add(bar, 1u, __ATOMIC_RELEASE,                        \
                             __HIP_MEMORY_SCOPE_AGENT);                        \
      ++barcount;                                                              \
      unsigned tgt = barcount * RG;                                            \
      while (__hip_atomic_load(bar, __ATOMIC_ACQUIRE,                          \
                               __HIP_MEMORY_SCOPE_AGENT) < tgt)                \
        __builtin_amdgcn_s_sleep(1);                                           \
    }                                                                          \
    __syncthreads();                                                           \
  } while (0)

  GRID_BARRIER();                          // all slices loaded, h zeroed

  for (int t = 0; t < T; ++t) {
    // ---- cache this lane's 64 h-values (32 uints) from the read buffer ----
    const unsigned int* hr = (const unsigned int*)(hbuf + (t & 1) * H_DIM);
    v2f hv[32];
#pragma unroll
    for (int j = 0; j < 8; ++j) {
      v4u hw = *(const v4u*)(hr + j * 128 + lane * 4);
#pragma unroll
      for (int q = 0; q < 4; ++q) {
        unsigned u = hw[q];
        v2f tmp = { bf16lo(u), bf16hi(u) };
        hv[j * 4 + q] = tmp;
      }
    }
    // ---- 8 rows per wave: LDS-resident dot products ----
#pragma unroll
    for (int rr = 0; rr < 8; ++rr) {
      const int row = wid * 8 + rr;
      const unsigned int* wu = (const unsigned int*)(lw + (size_t)row * ROWU16);
      v2f acc = { 0.0f, 0.0f };
#pragma unroll
      for (int j = 0; j < 8; ++j) {
        v4u wv = *(const v4u*)(wu + j * 128 + lane * 4);   // ds_load_b128
#pragma unroll
        for (int q = 0; q < 4; ++q) {
          unsigned u = wv[q];
          v2f wf = { bf16lo(u), bf16hi(u) };
          acc += wf * hv[j * 4 + q];                       // v_pk_fma_f32-able
        }
      }
      float s = acc[0] + acc[1];
#pragma unroll
      for (int off = 16; off > 0; off >>= 1) s += __shfl_xor(s, off, 32);
      if (lane == 0) zsum[row] = s;
    }
    __syncthreads();

    // ---- gate nonlinearities + state update (16 lanes) ----
    if (tid < RU) {
      const int gu = u0 + tid;
      const float* xp = xproj + (size_t)t * FOURH;
      float zi = zsum[tid]      + xp[gu];
      float zf = zsum[16 + tid] + xp[H_DIM + gu];
      float zg = zsum[32 + tid] + xp[2 * H_DIM + gu];
      float zo = zsum[48 + tid] + xp[3 * H_DIM + gu];
      float ig = 1.0f / (1.0f + __expf(-zi));
      float fg = 1.0f / (1.0f + __expf(-zf));
      float gg = tanhf(zg);
      float og = 1.0f / (1.0f + __expf(-zo));
      float c  = fg * cst[tid] + ig * gg;
      cst[tid] = c;
      float h  = og * tanhf(c);
      unsigned short hb = f32_to_bf16(h);
      hbuf[((t & 1) ^ 1) * H_DIM + gu] = hb;   // write buffer
      ys[(size_t)t * H_DIM + gu] = hb;
      hf32[gu] = h;
    }
    GRID_BARRIER();                            // publish h(t) to all WGs
  }
#undef GRID_BARRIER
}

// ---------------------------------------------------------------------------
// FC: out[o] = fc_b[o] + h . fc_w[o,:]   (one wave per output)
// ---------------------------------------------------------------------------
__global__ __launch_bounds__(256) void fc_kernel(const float* __restrict__ h,
                                                 const float* __restrict__ W,
                                                 const float* __restrict__ b,
                                                 float* __restrict__ out) {
  const int wave = (int)((blockIdx.x * blockDim.x + threadIdx.x) >> 5);
  const int lane = (int)(threadIdx.x & 31);
  if (wave >= O_DIM) return;
  const float* wr = W + (size_t)wave * H_DIM;
  float s = 0.0f;
  for (int k = lane; k < H_DIM; k += 32) s += h[k] * wr[k];
#pragma unroll
  for (int off = 16; off > 0; off >>= 1) s += __shfl_xor(s, off, 32);
  if (lane == 0) out[wave] = s + b[wave];
}

// ---------------------------------------------------------------------------
extern "C" void kernel_launch(void* const* d_in, const int* in_sizes, int n_in,
                              void* d_out, int out_size, void* d_ws, size_t ws_size,
                              hipStream_t stream) {
  (void)in_sizes; (void)n_in; (void)out_size; (void)ws_size;
  const float* x    = (const float*)d_in[0];
  const float* Wih0 = (const float*)d_in[1];
  const float* Whh0 = (const float*)d_in[2];
  const float* bih0 = (const float*)d_in[3];
  const float* bhh0 = (const float*)d_in[4];
  const float* Wih1 = (const float*)d_in[5];
  const float* Whh1 = (const float*)d_in[6];
  const float* bih1 = (const float*)d_in[7];
  const float* bhh1 = (const float*)d_in[8];
  const float* fc_w = (const float*)d_in[9];
  const float* fc_b = (const float*)d_in[10];

  // ---- workspace layout (all offsets 256B aligned) ----
  char* w = (char*)d_ws;
  size_t off = 0;
  unsigned*       bar  = (unsigned*)(w + off);        off += 256;
  unsigned short* hbuf = (unsigned short*)(w + off);  off += (size_t)2 * H_DIM * 2;
  float*          hf32 = (float*)(w + off);           off += (size_t)H_DIM * 4;
  unsigned short* xbf  = (unsigned short*)(w + off);  off += (size_t)T_SEQ * D_IN * 2;
  unsigned short* w0bf = (unsigned short*)(w + off);  off += (size_t)FOURH * D_IN * 2;
  unsigned short* w1bf = (unsigned short*)(w + off);  off += (size_t)FOURH * H_DIM * 2;
  unsigned short* ysbf = (unsigned short*)(w + off);  off += (size_t)T_SEQ * H_DIM * 2;
  float*          xprj = (float*)(w + off);           off += (size_t)T_SEQ * FOURH * 4;

  // bf16 copies of the GEMM operands
  f2bf_kernel<<<512,  256, 0, stream>>>(x,    xbf,  (size_t)T_SEQ * D_IN);
  f2bf_kernel<<<2048, 256, 0, stream>>>(Wih0, w0bf, (size_t)FOURH * D_IN);
  f2bf_kernel<<<8192, 256, 0, stream>>>(Wih1, w1bf, (size_t)FOURH * H_DIM);

  const int gemm_blocks = (T_SEQ / 16) * (FOURH / 16) / 8;   // 8 waves/block

  // ---- layer 0 ----
  gemm_bf16_wmma<<<gemm_blocks, 256, 0, stream>>>(xbf, D_IN, w0bf, D_IN,
                                                  bih0, bhh0, xprj,
                                                  T_SEQ, FOURH, D_IN);
  zero_bar_kernel<<<1, 1, 0, stream>>>(bar);
  lstm_recurrence<<<RG, RB, RSMEM, stream>>>(Whh0, xprj, ysbf, hbuf, hf32,
                                             bar, T_SEQ);

  // ---- layer 1 ----
  gemm_bf16_wmma<<<gemm_blocks, 256, 0, stream>>>(ysbf, H_DIM, w1bf, H_DIM,
                                                  bih1, bhh1, xprj,
                                                  T_SEQ, FOURH, H_DIM);
  zero_bar_kernel<<<1, 1, 0, stream>>>(bar);
  lstm_recurrence<<<RG, RB, RSMEM, stream>>>(Whh1, xprj, ysbf, hbuf, hf32,
                                             bar, T_SEQ);

  // ---- final FC from fp32 last hidden state ----
  fc_kernel<<<(O_DIM * 32) / 256, 256, 0, stream>>>(hf32, fc_w, fc_b,
                                                    (float*)d_out);
}
